// GPTLanguageModel_19963007992269
// MI455X (gfx1250) — matrix operand order repaired and checked by reading, not compiled, not run on hardware
//
#include <hip/hip_runtime.h>

// ---------------------------------------------------------------------------
// GPT forward for gfx1250 (MI455X): bf16 WMMA GEMMs + flash attention.
// B=4, T=1024, C=1024, H=16, D=64, L=6, V=32000, M = B*T = 4096.
// ---------------------------------------------------------------------------

typedef __attribute__((ext_vector_type(16))) __bf16 v16bf;
typedef __attribute__((ext_vector_type(8)))  float  f32x8;

struct __attribute__((aligned(16))) U4 { unsigned int x, y, z, w; };
struct __attribute__((aligned(16))) F4 { float x, y, z, w; };
struct __attribute__((aligned(8)))  U2 { unsigned int x, y; };

union Frag {            // 8 VGPRs = 16 bf16 values, A/B operand of wmma
    U4    q[2];
    v16bf v;
};

__device__ __forceinline__ unsigned short f2bf(float f) {
    union { float f; unsigned int u; } x; x.f = f;
    unsigned int r = x.u + 0x7FFFu + ((x.u >> 16) & 1u);   // RNE
    return (unsigned short)(r >> 16);
}

// ---------------------------------------------------------------------------
// Generic GEMM: out[M,N] = act(A[M,K] * W[K,N] + bias) + resid
// Tile 128x128x32, 256 threads (8 waves), wave computes 64x32 via 4x2 WMMAs.
// Double-buffered LDS ping-pong: global prefetch of slab k+1 overlaps the
// WMMA consumption of slab k; one barrier per K-step.
// ---------------------------------------------------------------------------
#define LDS_STRIDE 40   // 32 halfs + 8 pad -> conflict-free b128 frag loads

template<int BIAS, int RELU, int RESID>
__global__ __launch_bounds__(256)
void gemm_kernel(const float* __restrict__ A, const float* __restrict__ W,
                 const float* __restrict__ bias, const float* __restrict__ resid,
                 float* __restrict__ out, int M, int N, int K)
{
    __shared__ unsigned short As[2][128 * LDS_STRIDE];   // [m][k] row-major
    __shared__ unsigned short Bs[2][128 * LDS_STRIDE];   // [n][k] (K-transposed)

    const int tid  = threadIdx.x;
    const int lane = tid & 31;
    const int wave = tid >> 5;
    const int wm   = wave & 1;          // 0..1 : 64-row half
    const int wn   = wave >> 1;         // 0..3 : 32-col quarter
    const int m0   = blockIdx.y * 128;
    const int n0   = blockIdx.x * 128;

    f32x8 acc[4][2];
#pragma unroll
    for (int a = 0; a < 4; ++a)
#pragma unroll
        for (int b = 0; b < 2; ++b)
#pragma unroll
            for (int r = 0; r < 8; ++r) acc[a][b][r] = 0.f;

    const int a_k = (tid & 7) * 4;      // k sub-offset for A staging
    const int a_r = tid >> 3;           // row 0..31 (x4 steps)
    const int b_n = (tid & 31) * 4;     // n sub-offset for B staging
    const int b_k = tid >> 5;           // k row 0..7 (x4 steps)

    const int abase = (lane & 16) ? 8 : 0;    // A-frag K base
    const int bbase = (lane & 16) ? 8 : 0;    // B-frag K base

    F4 pa[4], pw[4];                    // prefetch registers

    auto load_slab = [&](int k0) {
#pragma unroll
        for (int i = 0; i < 4; ++i)
            pa[i] = *(const F4*)(A + (size_t)(m0 + a_r + 32 * i) * K + k0 + a_k);
#pragma unroll
        for (int i = 0; i < 4; ++i)
            pw[i] = *(const F4*)(W + (size_t)(k0 + b_k + 8 * i) * N + n0 + b_n);
    };
    auto store_slab = [&](int buf) {
#pragma unroll
        for (int i = 0; i < 4; ++i) {
            const int row = a_r + 32 * i;
            U2 p;
            p.x = (unsigned)f2bf(pa[i].x) | ((unsigned)f2bf(pa[i].y) << 16);
            p.y = (unsigned)f2bf(pa[i].z) | ((unsigned)f2bf(pa[i].w) << 16);
            *(U2*)(&As[buf][row * LDS_STRIDE + a_k]) = p;
        }
#pragma unroll
        for (int i = 0; i < 4; ++i) {
            const int kk = b_k + 8 * i;
            Bs[buf][(b_n + 0) * LDS_STRIDE + kk] = f2bf(pw[i].x);
            Bs[buf][(b_n + 1) * LDS_STRIDE + kk] = f2bf(pw[i].y);
            Bs[buf][(b_n + 2) * LDS_STRIDE + kk] = f2bf(pw[i].z);
            Bs[buf][(b_n + 3) * LDS_STRIDE + kk] = f2bf(pw[i].w);
        }
    };

    const int nk = K >> 5;
    load_slab(0);
    store_slab(0);

    for (int kt = 0; kt < nk; ++kt) {
        __syncthreads();                         // publishes buf kt&1
        if (kt + 1 < nk) load_slab((kt + 1) * 32);   // overlap with WMMAs

        const unsigned short* Ab = As[kt & 1];
        const unsigned short* Bb = Bs[kt & 1];
        Frag af[4], bf_[2];
#pragma unroll
        for (int a = 0; a < 4; ++a) {
            const int row = wm * 64 + a * 16 + (lane & 15);
            const unsigned short* p = &Ab[row * LDS_STRIDE + abase];
            af[a].q[0] = *(const U4*)p;          // K = abase .. abase+7
            af[a].q[1] = *(const U4*)(p + 16);   // K = abase+16 .. abase+23
        }
#pragma unroll
        for (int b = 0; b < 2; ++b) {
            const int col = wn * 32 + b * 16 + (lane & 15);
            const unsigned short* p = &Bb[col * LDS_STRIDE + bbase];
            bf_[b].q[0] = *(const U4*)p;         // K = bbase .. bbase+7
            bf_[b].q[1] = *(const U4*)(p + 16);  // K = bbase+16 .. bbase+23
        }
#pragma unroll
        for (int a = 0; a < 4; ++a)
#pragma unroll
            for (int b = 0; b < 2; ++b)
                acc[a][b] = __builtin_amdgcn_wmma_f32_16x16x32_bf16(
                    false, af[a].v, false, bf_[b].v, (short)0, acc[a][b], false, false);

        if (kt + 1 < nk) store_slab((kt + 1) & 1);
    }

    // ---- epilogue ----
    const int rowadd = (lane & 16) ? 8 : 0;
#pragma unroll
    for (int a = 0; a < 4; ++a) {
#pragma unroll
        for (int b = 0; b < 2; ++b) {
            const int col = n0 + wn * 32 + b * 16 + (lane & 15);
            float bv = 0.f;
            if (BIAS) bv = bias[col];
#pragma unroll
            for (int r = 0; r < 8; ++r) {
                const int row = m0 + wm * 64 + a * 16 + r + rowadd;
                float v = acc[a][b][r] + bv;
                if (RELU) v = fmaxf(v, 0.f);
                if (RESID) v += resid[(size_t)row * N + col];
                out[(size_t)row * N + col] = v;
            }
        }
    }
}

// ---------------------------------------------------------------------------
// Flash attention: grid (T/128, B*H). Q pre-scaled by 1/sqrt(D).
// q/k/v are bf16 in (B,H,T,D) layout; output written into (B,T,C) fp32.
// K tile staged with GLOBAL_LOAD_ASYNC_TO_LDS_B128 (ASYNCcnt path).
// ---------------------------------------------------------------------------
__global__ __launch_bounds__(256)
void attn_kernel(const unsigned short* __restrict__ Q,
                 const unsigned short* __restrict__ Kk,
                 const unsigned short* __restrict__ V,
                 float* __restrict__ O)
{
    __shared__ unsigned short Kt[32 * 80];       // [k][d], pad to 80 halfs
    __shared__ unsigned short Vt[64 * 40];       // [d][k], pad to 40 halfs
    __shared__ unsigned short Pt[8 * 16 * 40];   // per-wave P tile [16][32]+pad

    const int tid  = threadIdx.x;
    const int lane = tid & 31;
    const int wave = tid >> 5;
    const int bh   = blockIdx.y;                 // b*16 + h
    const int q0   = blockIdx.x * 128;
    const size_t base = (size_t)bh * 1024 * 64;

    // Q fragments (2 chunks of K=32 over D=64), loaded straight from global
    Frag aq[2];
    {
        const int t    = q0 + wave * 16 + (lane & 15);
        const int dsel = (lane & 16) ? 8 : 0;
        const unsigned short* qp = Q + base + (size_t)t * 64;
#pragma unroll
        for (int c = 0; c < 2; ++c) {
            aq[c].q[0] = *(const U4*)(qp + c * 32 + dsel);
            aq[c].q[1] = *(const U4*)(qp + c * 32 + dsel + 16);
        }
    }

    f32x8 o[4];
#pragma unroll
    for (int dt = 0; dt < 4; ++dt)
#pragma unroll
        for (int r = 0; r < 8; ++r) o[dt][r] = 0.f;
    float mrun[8], lrun[8];
#pragma unroll
    for (int r = 0; r < 8; ++r) { mrun[r] = -1e30f; lrun[r] = 0.f; }

    const int nkt  = q0 / 32 + 4;                // causal extent of this block
    const int st_r = tid >> 3;                   // staging: key row 0..31
    const int st_d = (tid & 7) * 8;              // staging: d offset
    unsigned short* Pw = Pt + wave * 16 * 40;
    const int bsel   = (lane & 16) ? 8 : 0;
    const int rowadd = (lane & 16) ? 8 : 0;

    const unsigned short* kbase = Kk + base;     // uniform -> SGPR pair
    const unsigned ldsK = (unsigned)(uintptr_t)&Kt[st_r * 80 + st_d];

    for (int kt = 0; kt < nkt; ++kt) {
        __syncthreads();
        {   // stage K tile [32][64] via async DMA to LDS; V^T tile manually
            const unsigned voff =
                (unsigned)(((kt * 32 + st_r) * 64 + st_d) * 2);   // bytes
            asm volatile("global_load_async_to_lds_b128 %0, %1, %2"
                         :: "v"(ldsK), "v"(voff), "s"(kbase) : "memory");
            U4 vv = *(const U4*)(V + base + (size_t)(kt * 32 + st_r) * 64 + st_d);
            const unsigned short* pv = (const unsigned short*)&vv;
#pragma unroll
            for (int j = 0; j < 8; ++j)
                Vt[(st_d + j) * 40 + st_r] = pv[j];
            asm volatile("s_wait_asynccnt 0x0" ::: "memory");
        }
        __syncthreads();

        // ---- S = Q K^T (two 16x16 tiles over 32 keys) ----
        f32x8 s[2];
#pragma unroll
        for (int j = 0; j < 2; ++j) {
#pragma unroll
            for (int r = 0; r < 8; ++r) s[j][r] = 0.f;
            const int krow = j * 16 + (lane & 15);
#pragma unroll
            for (int c = 0; c < 2; ++c) {
                Frag kf;
                const unsigned short* p = &Kt[krow * 80 + c * 32 + bsel];
                kf.q[0] = *(const U4*)p;
                kf.q[1] = *(const U4*)(p + 16);
                s[j] = __builtin_amdgcn_wmma_f32_16x16x32_bf16(
                    false, aq[c].v, false, kf.v, (short)0, s[j], false, false);
            }
        }

        // ---- causal mask + online softmax ----
        const int qrow0 = q0 + wave * 16 + rowadd;
        const int ncol0 = kt * 32 + (lane & 15);
#pragma unroll
        for (int r = 0; r < 8; ++r) {
            const int qr = qrow0 + r;
            float s0 = (ncol0      <= qr) ? s[0][r] : -1e30f;
            float s1 = (ncol0 + 16 <= qr) ? s[1][r] : -1e30f;
            float mx = fmaxf(s0, s1);
#pragma unroll
            for (int off = 1; off < 16; off <<= 1)
                mx = fmaxf(mx, __shfl_xor(mx, off, 32));
            const float mnew  = fmaxf(mrun[r], mx);
            const float alpha = __expf(mrun[r] - mnew);
            const float p0 = __expf(s0 - mnew);
            const float p1 = __expf(s1 - mnew);
            float ssum = p0 + p1;
#pragma unroll
            for (int off = 1; off < 16; off <<= 1)
                ssum += __shfl_xor(ssum, off, 32);
            lrun[r] = lrun[r] * alpha + ssum;
            mrun[r] = mnew;
#pragma unroll
            for (int dt = 0; dt < 4; ++dt) o[dt][r] *= alpha;
            // scatter P (C-layout) into per-wave LDS tile
            Pw[(r + rowadd) * 40 + (lane & 15)]      = f2bf(p0);
            Pw[(r + rowadd) * 40 + 16 + (lane & 15)] = f2bf(p1);
        }

        // ---- O += P V ----
        Frag ap;
        {
            const int pb = (lane & 16) ? 8 : 0;
            const unsigned short* p = Pw + (lane & 15) * 40 + pb;
            ap.q[0] = *(const U4*)p;
            ap.q[1] = *(const U4*)(p + 16);
        }
#pragma unroll
        for (int dt = 0; dt < 4; ++dt) {
            Frag vf;
            const unsigned short* p = &Vt[(dt * 16 + (lane & 15)) * 40 + bsel];
            vf.q[0] = *(const U4*)p;
            vf.q[1] = *(const U4*)(p + 16);
            o[dt] = __builtin_amdgcn_wmma_f32_16x16x32_bf16(
                false, ap.v, false, vf.v, (short)0, o[dt], false, false);
        }
    }

    // ---- write O / l into (B,T,C) buffer ----
    const int b = bh >> 4, h = bh & 15;
#pragma unroll
    for (int r = 0; r < 8; ++r) {
        const int t = q0 + wave * 16 + r + rowadd;
        const float inv = 1.f / lrun[r];
        float* op = O + ((size_t)(b * 1024 + t)) * 1024 + h * 64;
#pragma unroll
        for (int dt = 0; dt < 4; ++dt)
            op[dt * 16 + (lane & 15)] = o[dt][r] * inv;
    }
}

// ---------------------------------------------------------------------------
// Helpers: embedding, layernorm, qkv split/cast
// ---------------------------------------------------------------------------
__global__ __launch_bounds__(256)
void embed_kernel(const int* __restrict__ idx, const float* __restrict__ tok,
                  const float* __restrict__ pos, float* __restrict__ x)
{
    const int row = blockIdx.x;          // b*T + t
    const int t   = row & 1023;
    const int tk  = idx[row];
#pragma unroll
    for (int i = 0; i < 4; ++i) {
        const int c = threadIdx.x + 256 * i;
        x[(size_t)row * 1024 + c] = tok[(size_t)tk * 1024 + c] + pos[(size_t)t * 1024 + c];
    }
}

__global__ __launch_bounds__(256)
void layernorm_kernel(const float* __restrict__ x, const float* __restrict__ w,
                      const float* __restrict__ b, float* __restrict__ out)
{
    __shared__ float red[256];
    const int row = blockIdx.x, tid = threadIdx.x;
    const float* xr = x + (size_t)row * 1024;
    float v[4];
    float s = 0.f;
#pragma unroll
    for (int i = 0; i < 4; ++i) { v[i] = xr[tid + 256 * i]; s += v[i]; }
    red[tid] = s; __syncthreads();
    for (int off = 128; off > 0; off >>= 1) {
        if (tid < off) red[tid] += red[tid + off];
        __syncthreads();
    }
    const float mu = red[0] * (1.f / 1024.f);
    __syncthreads();
    s = 0.f;
#pragma unroll
    for (int i = 0; i < 4; ++i) { const float d = v[i] - mu; s += d * d; }
    red[tid] = s; __syncthreads();
    for (int off = 128; off > 0; off >>= 1) {
        if (tid < off) red[tid] += red[tid + off];
        __syncthreads();
    }
    const float rstd = rsqrtf(red[0] * (1.f / 1024.f) + 1e-5f);
    float* orow = out + (size_t)row * 1024;
#pragma unroll
    for (int i = 0; i < 4; ++i) {
        const int c = tid + 256 * i;
        orow[c] = (v[i] - mu) * rstd * w[c] + b[c];
    }
}

__global__ __launch_bounds__(256)
void split_qkv_kernel(const float* __restrict__ qkv, unsigned short* __restrict__ q,
                      unsigned short* __restrict__ k, unsigned short* __restrict__ v)
{
    const int row = blockIdx.x;                  // b*T + t
    const int b = row >> 10, t = row & 1023;
    const float* src = qkv + (size_t)row * 3072;
    for (int e = threadIdx.x; e < 3072; e += 256) {
        const int sec = e >> 10, cc = e & 1023;
        const int h = cc >> 6, d = cc & 63;
        const size_t di = ((size_t)(b * 16 + h) * 1024 + t) * 64 + d;
        const float val = src[e];
        if (sec == 0)      q[di] = f2bf(val * 0.125f);   // 1/sqrt(64)
        else if (sec == 1) k[di] = f2bf(val);
        else               v[di] = f2bf(val);
    }
}

// ---------------------------------------------------------------------------
// Orchestration
// ---------------------------------------------------------------------------
extern "C" void kernel_launch(void* const* d_in, const int* in_sizes, int n_in,
                              void* d_out, int out_size, void* d_ws, size_t ws_size,
                              hipStream_t stream)
{
    (void)in_sizes; (void)n_in; (void)out_size; (void)ws_size;

    const int*   idx   = (const int*)  d_in[0];
    const float* tok   = (const float*)d_in[1];
    const float* pos   = (const float*)d_in[2];
    const float* qkv_w = (const float*)d_in[3];
    const float* ow    = (const float*)d_in[4];
    const float* ob    = (const float*)d_in[5];
    const float* f1w   = (const float*)d_in[6];
    const float* f1b   = (const float*)d_in[7];
    const float* f2w   = (const float*)d_in[8];
    const float* f2b   = (const float*)d_in[9];
    const float* ln1w  = (const float*)d_in[10];
    const float* ln1b  = (const float*)d_in[11];
    const float* ln2w  = (const float*)d_in[12];
    const float* ln2b  = (const float*)d_in[13];
    const float* lnfw  = (const float*)d_in[14];
    const float* lnfb  = (const float*)d_in[15];
    const float* headw = (const float*)d_in[16];

    char* ws = (char*)d_ws;
    float*          X   = (float*)(ws + ((size_t)0   << 20));  // 16 MB (B,T,C)
    float*          Hb  = (float*)(ws + ((size_t)16  << 20));  // 16 MB LN out
    float*          QKV = (float*)(ws + ((size_t)32  << 20));  // 48 MB
    float*          HID = (float*)(ws + ((size_t)80  << 20));  // 64 MB
    float*          AO  = (float*)(ws + ((size_t)144 << 20));  // 16 MB
    unsigned short* QB  = (unsigned short*)(ws + ((size_t)160 << 20));  // 8 MB
    unsigned short* KB  = (unsigned short*)(ws + ((size_t)168 << 20));  // 8 MB
    unsigned short* VB  = (unsigned short*)(ws + ((size_t)176 << 20));  // 8 MB

    const dim3 blk(256);
    const int  Mrows = 4096;

    embed_kernel<<<Mrows, blk, 0, stream>>>(idx, tok, pos, X);

    for (int l = 0; l < 6; ++l) {
        layernorm_kernel<<<Mrows, blk, 0, stream>>>(X, ln1w + l * 1024, ln1b + l * 1024, Hb);
        gemm_kernel<0,0,0><<<dim3(24, 32), blk, 0, stream>>>(
            Hb, qkv_w + (size_t)l * 1024 * 3072, nullptr, nullptr, QKV, 4096, 3072, 1024);
        split_qkv_kernel<<<Mrows, blk, 0, stream>>>(QKV, QB, KB, VB);
        attn_kernel<<<dim3(8, 64), blk, 0, stream>>>(QB, KB, VB, AO);
        gemm_kernel<1,0,1><<<dim3(8, 32), blk, 0, stream>>>(
            AO, ow + (size_t)l * 1024 * 1024, ob + l * 1024, X, X, 4096, 1024, 1024);
        layernorm_kernel<<<Mrows, blk, 0, stream>>>(X, ln2w + l * 1024, ln2b + l * 1024, Hb);
        gemm_kernel<1,1,0><<<dim3(32, 32), blk, 0, stream>>>(
            Hb, f1w + (size_t)l * 1024 * 4096, f1b + l * 4096, nullptr, HID, 4096, 4096, 1024);
        gemm_kernel<1,0,1><<<dim3(8, 32), blk, 0, stream>>>(
            HID, f2w + (size_t)l * 4096 * 1024, f2b + l * 1024, X, X, 4096, 1024, 4096);
    }

    layernorm_kernel<<<Mrows, blk, 0, stream>>>(X, lnfw, lnfb, Hb);
    gemm_kernel<0,0,0><<<dim3(250, 32), blk, 0, stream>>>(
        Hb, headw, nullptr, nullptr, (float*)d_out, 4096, 32000, 1024);
}
